// SparseData_Transformer_MoE_38182259261874
// MI455X (gfx1250) — matrix operand order repaired and checked by reading, not compiled, not run on hardware
//
#include <hip/hip_runtime.h>

// ---------------- types & WMMA helpers ----------------
typedef __attribute__((ext_vector_type(8)))  float  v8f;
typedef __attribute__((ext_vector_type(16))) __bf16 v16bf;
typedef __attribute__((ext_vector_type(8)))  __bf16 v8bf;
typedef __attribute__((ext_vector_type(4)))  int    i32x4;

static __device__ __forceinline__ v16bf cat8(v8bf lo, v8bf hi) {
  return __builtin_shufflevector(lo, hi, 0,1,2,3,4,5,6,7,8,9,10,11,12,13,14,15);
}
static __device__ __forceinline__ v8f wmma_bf16(v16bf a, v16bf b, v8f c) {
  // D = A(16x32 bf16) x B(32x16 bf16) + C(16x16 f32)
  return __builtin_amdgcn_wmma_f32_16x16x32_bf16(false, a, false, b, (short)0, c, false, false);
}

// ---------------- async global->LDS copy (CDNA5), with safe fallback ----------------
#if defined(__has_builtin)
#  if __has_builtin(__builtin_amdgcn_global_load_async_to_lds_b128)
#    define HAVE_ASYNC_LDS 1
#  endif
#endif
#ifndef HAVE_ASYNC_LDS
#  define HAVE_ASYNC_LDS 0
#endif

static __device__ __forceinline__ void cp16(__bf16* dst, const __bf16* src) {
#if HAVE_ASYNC_LDS
  __builtin_amdgcn_global_load_async_to_lds_b128(
      (__attribute__((address_space(1))) i32x4*)src,
      (__attribute__((address_space(3))) i32x4*)dst, 0, 0);
#else
  *(v8bf*)dst = *(const v8bf*)src;
#endif
}
static __device__ __forceinline__ void async_join() {
#if HAVE_ASYNC_LDS
#  if __has_builtin(__builtin_amdgcn_s_wait_asynccnt)
  __builtin_amdgcn_s_wait_asynccnt(0);
#  else
  asm volatile("s_wait_asynccnt 0x0" ::: "memory");
#  endif
#endif
}

#define BB  4
#define SS  2048
#define DD  512
#define HH  8
#define DH  64
#define EE  8
#define HID 1365
#define HIDP 1376
#define CAP 1024
#define NTOK (BB*SS)        // 8192
#define ROWS_E (BB*CAP)     // 4096

// ---------------- generic bf16 WMMA GEMM ----------------
// C[M,NP] = act(A[M,K] * B + bias) (+Src). A: bf16 [M][K].
// TRANSB: B bf16 [N][K] (ld=K); else B bf16 [K][N] (ld=N, rows valid < KB).
// Cols in [N,NP) written as 0 (K-padding for the consumer GEMM).
template<int TRANSB, int ACT, int ADDSRC, int OUTBF>
__global__ __launch_bounds__(256) void gemm_bf16_wmma(
    const __bf16* __restrict__ A, const __bf16* __restrict__ Bg,
    const float* __restrict__ bias, const float* __restrict__ Src,
    void* __restrict__ CoutV, int M, int N, int K, int KB, int NP,
    long long aB, long long bB, long long biasB, long long cB)
{
  __shared__ __attribute__((aligned(64))) __bf16 As[128][32];
  __shared__ __attribute__((aligned(64))) __bf16 Bs[128][32];   // [n][k]
  int z = blockIdx.z;
  A += (long long)z * aB; Bg += (long long)z * bB; bias += (long long)z * biasB;
  __bf16* Cb = (__bf16*)CoutV + (long long)z * cB;
  float*  Cf = (float*)CoutV  + (long long)z * cB;
  int tid = threadIdx.x, lane = tid & 31, wave = tid >> 5;
  int wm = wave & 3, wn = wave >> 2;         // 4x2 wave grid (32x64 per wave)
  int l16 = lane & 15, lh = lane >> 4;
  int row0 = blockIdx.y * 128, col0 = blockIdx.x * 128;
  v8f acc[2][4];
#pragma unroll
  for (int i = 0; i < 2; ++i)
#pragma unroll
    for (int j = 0; j < 4; ++j)
#pragma unroll
      for (int r = 0; r < 8; ++r) acc[i][j][r] = 0.f;

  for (int kk = 0; kk < K; kk += 32) {
    // A tile: 128x32 bf16, 512 16B-chunks / 256 thr = 2 each (async -> LDS)
#pragma unroll
    for (int it = 0; it < 2; ++it) {
      int idx = tid + it * 256;
      int r = idx >> 2, c = (idx & 3) * 8;
      int gr = row0 + r;
      if (gr < M) {
        const __bf16* src = &A[(long long)gr * K + kk + c];
        cp16(&As[r][c], src);
        if (kk + 32 < K) __builtin_prefetch(src + 32, 0, 0);
      } else {
        v8bf zz; for (int q = 0; q < 8; ++q) zz[q] = (__bf16)0.f;
        *(v8bf*)&As[r][c] = zz;
      }
    }
    if (TRANSB) {
#pragma unroll
      for (int it = 0; it < 2; ++it) {
        int idx = tid + it * 256;
        int r = idx >> 2, c = (idx & 3) * 8;
        int gn = col0 + r;
        if (gn < N) {
          cp16(&Bs[r][c], &Bg[(long long)gn * K + kk + c]);
        } else {
          v8bf zz; for (int q = 0; q < 8; ++q) zz[q] = (__bf16)0.f;
          *(v8bf*)&Bs[r][c] = zz;
        }
      }
    } else {
      // transpose staging: Bs[n][k] = B[k][n]; also zero-fills k >= KB pad rows
#pragma unroll
      for (int it = 0; it < 16; ++it) {
        int idx = tid + it * 256;          // 4096 scalars
        int kr = idx >> 7, nc = idx & 127;
        int gk = kk + kr, gn = col0 + nc;
        __bf16 v = (__bf16)0.f;
        if (gk < KB && gn < N) v = Bg[(long long)gk * N + gn];
        Bs[nc][kr] = v;
      }
    }
    async_join();
    __syncthreads();
    v16bf afr[2], bfr[4];
#pragma unroll
    for (int tm = 0; tm < 2; ++tm) {
      const __bf16* ap = &As[wm * 32 + tm * 16 + l16][lh * 16];
      afr[tm] = cat8(*(const v8bf*)ap, *(const v8bf*)(ap + 8));
    }
#pragma unroll
    for (int tn = 0; tn < 4; ++tn) {
      const __bf16* bp = &Bs[wn * 64 + tn * 16 + l16][lh * 16];
      bfr[tn] = *(const v16bf*)bp;
    }
#pragma unroll
    for (int tm = 0; tm < 2; ++tm)
#pragma unroll
      for (int tn = 0; tn < 4; ++tn)
        acc[tm][tn] = wmma_bf16(afr[tm], bfr[tn], acc[tm][tn]);
    __syncthreads();
  }
#pragma unroll
  for (int tm = 0; tm < 2; ++tm)
#pragma unroll
    for (int tn = 0; tn < 4; ++tn)
#pragma unroll
      for (int r = 0; r < 8; ++r) {
        int grow = row0 + wm * 32 + tm * 16 + lh * 8 + r;
        int gcol = col0 + wn * 64 + tn * 16 + l16;
        if (grow >= M) continue;
        long long co = (long long)grow * NP + gcol;
        if (gcol < N) {
          float v = acc[tm][tn][r] + bias[gcol];
          if (ACT == 1) v = (v > 0.f) ? v : 0.01f * v;   // LeakyReLU
          if (ADDSRC) v += Src[co];
          if (OUTBF) Cb[co] = (__bf16)v; else Cf[co] = v;
        } else if (gcol < NP) {
          if (OUTBF) Cb[co] = (__bf16)0.f; else Cf[co] = 0.f;
        }
      }
}

// ---------------- flash attention (bf16 WMMA, online softmax) ----------------
__global__ __launch_bounds__(256) void attn_kernel(const __bf16* __restrict__ qkv,
                                                   __bf16* __restrict__ O)
{
  __shared__ __attribute__((aligned(64))) __bf16 Qs[128][64];
  __shared__ __attribute__((aligned(64))) __bf16 Ks[64][64];
  __shared__ __attribute__((aligned(64))) __bf16 Vs[64][64];     // [d][key]
  __shared__ __attribute__((aligned(64))) __bf16 Ps[8][16][64];  // per-wave P tile
  int b = blockIdx.z, h = blockIdx.y, q0 = blockIdx.x * 128;
  int tid = threadIdx.x, lane = tid & 31, wave = tid >> 5;
  int l16 = lane & 15, lh = lane >> 4;
  const long long ld = 3 * DD;
  const __bf16* qb = qkv + ((long long)b * SS + q0) * ld + h * DH;
  const __bf16* kb = qkv + ((long long)b * SS) * ld + DD + h * DH;
  const __bf16* vb = qkv + ((long long)b * SS) * ld + 2 * DD + h * DH;
  // stage Q (128x64 bf16): 1024 chunks / 256 thr = 4 each (async)
#pragma unroll
  for (int it = 0; it < 4; ++it) {
    int idx = tid + it * 256;
    int r = idx >> 3, c = (idx & 7) * 8;
    cp16(&Qs[r][c], &qb[(long long)r * ld + c]);
  }
  async_join();
  __syncthreads();
  v16bf qa[2];
#pragma unroll
  for (int f = 0; f < 2; ++f) {
    const __bf16* ap = &Qs[wave * 16 + l16][f * 32 + lh * 16];
    qa[f] = cat8(*(const v8bf*)ap, *(const v8bf*)(ap + 8));
  }
  float m[8], lsum[8]; v8f o[4];
#pragma unroll
  for (int r = 0; r < 8; ++r) { m[r] = -1e30f; lsum[r] = 0.f; }
#pragma unroll
  for (int j = 0; j < 4; ++j)
#pragma unroll
    for (int r = 0; r < 8; ++r) o[j][r] = 0.f;

  for (int kt = 0; kt < SS; kt += 64) {
    // K tile: 64x64, 512 chunks -> async; V tile: transposed manual staging
#pragma unroll
    for (int it = 0; it < 2; ++it) {
      int idx = tid + it * 256;
      int r = idx >> 3, c = (idx & 7) * 8;
      cp16(&Ks[r][c], &kb[(long long)(kt + r) * ld + c]);
      v8bf vv = *(const v8bf*)&vb[(long long)(kt + r) * ld + c];
#pragma unroll
      for (int q = 0; q < 8; ++q) Vs[c + q][r] = vv[q];
    }
    async_join();
    __syncthreads();
    v8f s[4];
#pragma unroll
    for (int j = 0; j < 4; ++j)
#pragma unroll
      for (int r = 0; r < 8; ++r) s[j][r] = 0.f;
#pragma unroll
    for (int j = 0; j < 4; ++j)
#pragma unroll
      for (int f = 0; f < 2; ++f) {
        const __bf16* bp = &Ks[j * 16 + l16][f * 32 + lh * 16];
        s[j] = wmma_bf16(qa[f], *(const v16bf*)bp, s[j]);
      }
    float mt[8];
#pragma unroll
    for (int r = 0; r < 8; ++r) {
#pragma unroll
      for (int j = 0; j < 4; ++j) s[j][r] *= 0.125f;            // 1/sqrt(64)
      float v = fmaxf(fmaxf(s[0][r], s[1][r]), fmaxf(s[2][r], s[3][r]));
      v = fmaxf(v, __shfl_xor(v, 1)); v = fmaxf(v, __shfl_xor(v, 2));
      v = fmaxf(v, __shfl_xor(v, 4)); v = fmaxf(v, __shfl_xor(v, 8));
      mt[r] = v;
    }
#pragma unroll
    for (int r = 0; r < 8; ++r) {
      float mn = fmaxf(m[r], mt[r]);
      float al = __expf(m[r] - mn);
      m[r] = mn; lsum[r] *= al;
#pragma unroll
      for (int j = 0; j < 4; ++j) o[j][r] *= al;
    }
#pragma unroll
    for (int r = 0; r < 8; ++r) {
      float rs = 0.f;
#pragma unroll
      for (int j = 0; j < 4; ++j) {
        float p = __expf(s[j][r] - m[r]);
        rs += p;
        Ps[wave][lh * 8 + r][j * 16 + l16] = (__bf16)p;         // C-layout -> LDS
      }
      rs += __shfl_xor(rs, 1); rs += __shfl_xor(rs, 2);
      rs += __shfl_xor(rs, 4); rs += __shfl_xor(rs, 8);
      lsum[r] += rs;
    }
    // P (A-layout via per-wave LDS bounce) @ V
#pragma unroll
    for (int f = 0; f < 2; ++f) {
      const __bf16* ap = &Ps[wave][l16][f * 32 + lh * 16];
      v16bf pa = cat8(*(const v8bf*)ap, *(const v8bf*)(ap + 8));
#pragma unroll
      for (int j = 0; j < 4; ++j) {
        const __bf16* bp = &Vs[j * 16 + l16][f * 32 + lh * 16];
        o[j] = wmma_bf16(pa, *(const v16bf*)bp, o[j]);
      }
    }
    __syncthreads();
  }
  __bf16* ob = O + ((long long)b * SS + q0 + wave * 16) * DD + h * DH;
#pragma unroll
  for (int r = 0; r < 8; ++r) {
    float inv = 1.0f / lsum[r];
    int row = lh * 8 + r;
#pragma unroll
    for (int j = 0; j < 4; ++j)
      ob[(long long)row * DD + j * 16 + l16] = (__bf16)(o[j][r] * inv);
  }
}

// ---------------- gating: logits, softmax, top-2, loss accumulators ----------------
__global__ __launch_bounds__(256) void gate_kernel(const float* __restrict__ X,
    const float* __restrict__ Wg, int* gi0, int* gi1,
    float* gv0, float* gv1, float* use2, float* accum)
{
  int tok = blockIdx.x * 8 + (threadIdx.x >> 5);
  int lane = threadIdx.x & 31;
  const float* x = X + (long long)tok * DD;
  float a[EE];
#pragma unroll
  for (int e = 0; e < EE; ++e) a[e] = 0.f;
  for (int d = lane; d < DD; d += 32) {
    float xv = x[d];
    const float* w = Wg + (long long)d * EE;
#pragma unroll
    for (int e = 0; e < EE; ++e) a[e] += xv * w[e];
  }
#pragma unroll
  for (int e = 0; e < EE; ++e)
    for (int msk = 1; msk < 32; msk <<= 1) a[e] += __shfl_xor(a[e], msk);
  float mx = a[0];
#pragma unroll
  for (int e = 1; e < EE; ++e) mx = fmaxf(mx, a[e]);
  float p[EE], ssum = 0.f;
#pragma unroll
  for (int e = 0; e < EE; ++e) { p[e] = __expf(a[e] - mx); ssum += p[e]; }
  float lse = mx + __logf(ssum);
  float inv = 1.f / ssum;
#pragma unroll
  for (int e = 0; e < EE; ++e) p[e] *= inv;
  int i0 = 0;
#pragma unroll
  for (int e = 1; e < EE; ++e) if (p[e] > p[i0]) i0 = e;
  int i1 = (i0 == 0) ? 1 : 0;
#pragma unroll
  for (int e = 0; e < EE; ++e) if (e != i0 && p[e] > p[i1]) i1 = e;
  if (lane == 0) {
    gi0[tok] = i0; gi1[tok] = i1;
    gv0[tok] = p[i0]; gv1[tok] = p[i1];
    use2[tok] = (p[i1] > 0.2f) ? 1.f : 0.f;
    atomicAdd(&accum[i0], 1.f);                    // density (top1 one-hot)
#pragma unroll
    for (int e = 0; e < EE; ++e) atomicAdd(&accum[8 + e], p[e]);  // proxy
    atomicAdd(&accum[16], lse * lse);              // z-loss
  }
}

// ---------------- capacity scan (serial per batch row) ----------------
__global__ void scan_kernel(const int* gi0, const int* gi1, const float* gv0,
                            const float* gv1, const float* use2,
                            int* slot, float* keep, float* wcb)
{
  if (threadIdx.x != 0) return;
  int b = blockIdx.x;
  int cnt[EE];
  for (int e = 0; e < EE; ++e) cnt[e] = 0;
  for (int t = 0; t < 2 * SS; ++t) {
    int s = t & (SS - 1);
    int tok = b * SS + s;
    int e; float assigned, gw;
    if (t < SS) { e = gi0[tok]; assigned = 1.f; gw = gv0[tok]; }
    else        { float u = use2[tok]; e = gi1[tok]; assigned = u; gw = gv1[tok] * u; }
    int sl = CAP; float kp = 0.f;
    if (assigned > 0.f) {
      int ppos = cnt[e]++;
      if (ppos < CAP) { sl = ppos; kp = 1.f; }
    }
    long long off = (long long)b * (2 * SS) + t;
    slot[off] = sl; keep[off] = kp; wcb[off] = gw * kp;
  }
}

// ---------------- dispatch scatter (one wave per token-assignment) ----------------
__global__ __launch_bounds__(256) void dispatch_kernel(const float* __restrict__ X,
    const int* gi0, const int* gi1, const float* keep, const int* slot,
    float* __restrict__ xe)
{
  int idx = blockIdx.x * 8 + (threadIdx.x >> 5);
  int lane = threadIdx.x & 31;
  int b = idx >> 12, t = idx & 4095;
  long long off = (long long)b * (2 * SS) + t;
  if (keep[off] == 0.f) return;
  int s = t & (SS - 1);
  int tok = b * SS + s;
  int e = (t < SS) ? gi0[tok] : gi1[tok];
  const float* src = X + (long long)tok * DD;
  float* dst = xe + ((long long)e * ROWS_E + b * CAP + slot[off]) * DD;
#pragma unroll
  for (int it = 0; it < 4; ++it) {
    int d = lane * 4 + it * 128;
    *(float4*)&dst[d] = *(const float4*)&src[d];
  }
}

// ---------------- per-row LayerNorm: xe (f32) -> xn (bf16) ----------------
__global__ __launch_bounds__(256) void ln_kernel(const float* __restrict__ xe,
    const float* __restrict__ g, const float* __restrict__ bta,
    __bf16* __restrict__ xn)
{
  int row = blockIdx.x * 8 + (threadIdx.x >> 5);
  int lane = threadIdx.x & 31;
  int e = row >> 12;                     // 4096 rows per expert
  const float* x = xe + (long long)row * DD;
  float v[16], s = 0.f, s2 = 0.f;
#pragma unroll
  for (int i = 0; i < 16; ++i) { v[i] = x[lane + i * 32]; s += v[i]; s2 += v[i] * v[i]; }
  for (int msk = 1; msk < 32; msk <<= 1) { s += __shfl_xor(s, msk); s2 += __shfl_xor(s2, msk); }
  float mu = s * (1.f / DD);
  float var = s2 * (1.f / DD) - mu * mu;
  float rs = rsqrtf(var + 1e-5f);
  const float* gg = g + (long long)e * DD;
  const float* bb = bta + (long long)e * DD;
  __bf16* out = xn + (long long)row * DD;
#pragma unroll
  for (int i = 0; i < 16; ++i) {
    int d = lane + i * 32;
    out[d] = (__bf16)((v[i] - mu) * rs * gg[d] + bb[d]);
  }
}

// ---------------- combine: out = resid + sum_j w_j * y[e_j, slot_j] ----------------
__global__ __launch_bounds__(256) void combine_kernel(const float* __restrict__ R,
    const float* __restrict__ ye, const int* gi0, const int* gi1,
    const float* keep, const int* slot, const float* wcb, float* __restrict__ out)
{
  int tok = blockIdx.x * 8 + (threadIdx.x >> 5);
  int lane = threadIdx.x & 31;
  int b = tok >> 11, s = tok & (SS - 1);
  float acc[16];
  const float* r = R + (long long)tok * DD;
#pragma unroll
  for (int i = 0; i < 16; ++i) acc[i] = r[lane + i * 32];
  for (int j = 0; j < 2; ++j) {
    long long bt = (long long)b * (2 * SS) + j * SS + s;
    if (keep[bt] > 0.f) {
      int e = j ? gi1[tok] : gi0[tok];
      float w = wcb[bt];
      const float* y = ye + ((long long)e * ROWS_E + b * CAP + slot[bt]) * DD;
#pragma unroll
      for (int i = 0; i < 16; ++i) acc[i] += w * y[lane + i * 32];
    }
  }
  float* o = out + (long long)tok * DD;
#pragma unroll
  for (int i = 0; i < 16; ++i) o[lane + i * 32] = acc[i];
}

// ---------------- misc ----------------
__global__ void zero_kernel(float* p, long long n) {
  long long i = (long long)blockIdx.x * blockDim.x + threadIdx.x;
  long long st = (long long)gridDim.x * blockDim.x;
  for (; i < n; i += st) p[i] = 0.f;
}

__global__ void cvt_bf16_kernel(const float* __restrict__ s, __bf16* __restrict__ d,
                                long long n) {
  long long i = (long long)blockIdx.x * blockDim.x + threadIdx.x;
  long long st = (long long)gridDim.x * blockDim.x;
  for (; i < n; i += st) d[i] = (__bf16)s[i];
}

__global__ void finalize_kernel(const float* accum, float* tail) {
  if (threadIdx.x == 0 && blockIdx.x == 0) {
    float bal = 0.f;
    for (int e = 0; e < EE; ++e)
      bal += (accum[e] / (float)NTOK) * (accum[8 + e] / (float)NTOK);
    bal *= (float)EE;
    float z = accum[16] / (float)NTOK;
    tail[0] = 0.01f * bal + 0.001f * z;   // total
    tail[1] = bal;                        // balance
    tail[2] = z;                          // zloss
  }
}

// ---------------- launch ----------------
extern "C" void kernel_launch(void* const* d_in, const int* in_sizes, int n_in,
                              void* d_out, int out_size, void* d_ws, size_t ws_size,
                              hipStream_t stream) {
  (void)in_sizes; (void)n_in; (void)out_size; (void)ws_size;
  const float* x    = (const float*)d_in[0];
  const float* Wqkv = (const float*)d_in[1];
  const float* bqkv = (const float*)d_in[2];
  const float* Wo   = (const float*)d_in[3];
  const float* bo   = (const float*)d_in[4];
  const float* Wg   = (const float*)d_in[5];
  const float* ln_g = (const float*)d_in[6];
  const float* ln_b = (const float*)d_in[7];
  const float* W1   = (const float*)d_in[8];
  const float* b1   = (const float*)d_in[9];
  const float* W2   = (const float*)d_in[10];
  const float* b2   = (const float*)d_in[11];
  float* out = (float*)d_out;

  char* base = (char*)d_ws;
  size_t off = 0;
  auto alloc = [&](size_t bytes) -> void* {
    void* p = base + off;
    off = (off + bytes + 255) & ~(size_t)255;
    return p;
  };
  __bf16* xb    = (__bf16*)alloc((size_t)NTOK * DD * 2);          // x bf16
  __bf16* Wqkvb = (__bf16*)alloc((size_t)3 * DD * DD * 2);
  __bf16* Wob   = (__bf16*)alloc((size_t)DD * DD * 2);
  __bf16* W1b   = (__bf16*)alloc((size_t)EE * DD * HID * 2);
  __bf16* W2b   = (__bf16*)alloc((size_t)EE * HID * DD * 2);
  __bf16* qkvb  = (__bf16*)alloc((size_t)NTOK * 3 * DD * 2);
  __bf16* oatb  = (__bf16*)alloc((size_t)NTOK * DD * 2);
  float*  resid = (float*)alloc((size_t)NTOK * DD * 4);
  float*  xe    = (float*)alloc((size_t)EE * ROWS_E * DD * 4);
  __bf16* xnb   = (__bf16*)alloc((size_t)EE * ROWS_E * DD * 2);
  __bf16* hb    = (__bf16*)alloc((size_t)EE * ROWS_E * HIDP * 2);
  float*  ye    = (float*)alloc((size_t)EE * ROWS_E * DD * 4);
  float*  gv0   = (float*)alloc((size_t)NTOK * 4);
  float*  gv1   = (float*)alloc((size_t)NTOK * 4);
  float*  use2  = (float*)alloc((size_t)NTOK * 4);
  float*  keep  = (float*)alloc((size_t)2 * NTOK * 4);
  float*  wcb   = (float*)alloc((size_t)2 * NTOK * 4);
  float*  accum = (float*)alloc(32 * 4);
  int*    gi0   = (int*)alloc((size_t)NTOK * 4);
  int*    gi1   = (int*)alloc((size_t)NTOK * 4);
  int*    slot  = (int*)alloc((size_t)2 * NTOK * 4);

  dim3 blk(256);
  // bf16 conversions (once per call)
  cvt_bf16_kernel<<<1024, blk, 0, stream>>>(x,    xb,    (long long)NTOK * DD);
  cvt_bf16_kernel<<<512,  blk, 0, stream>>>(Wqkv, Wqkvb, (long long)3 * DD * DD);
  cvt_bf16_kernel<<<256,  blk, 0, stream>>>(Wo,   Wob,   (long long)DD * DD);
  cvt_bf16_kernel<<<1024, blk, 0, stream>>>(W1,   W1b,   (long long)EE * DD * HID);
  cvt_bf16_kernel<<<1024, blk, 0, stream>>>(W2,   W2b,   (long long)EE * HID * DD);
  zero_kernel<<<4096, blk, 0, stream>>>(xe, (long long)EE * ROWS_E * DD);
  zero_kernel<<<1, blk, 0, stream>>>(accum, 32);

  // qkv = x @ Wqkv.T + bqkv  -> bf16   [8192,1536]
  gemm_bf16_wmma<1,0,0,1><<<dim3(12,64,1), blk, 0, stream>>>(
      xb, Wqkvb, bqkv, nullptr, qkvb, NTOK, 3*DD, DD, DD, 3*DD, 0,0,0,0);

  // attention -> oat (bf16)
  attn_kernel<<<dim3(SS/128, HH, BB), blk, 0, stream>>>(qkvb, oatb);

  // resid = oat @ Wo.T + bo + x   (f32 residual stream)
  gemm_bf16_wmma<1,0,1,0><<<dim3(4,64,1), blk, 0, stream>>>(
      oatb, Wob, bo, x, resid, NTOK, DD, DD, DD, DD, 0,0,0,0);

  // gating + dispatch
  gate_kernel<<<NTOK/8, blk, 0, stream>>>(resid, Wg, gi0, gi1, gv0, gv1, use2, accum);
  scan_kernel<<<BB, 32, 0, stream>>>(gi0, gi1, gv0, gv1, use2, slot, keep, wcb);
  dispatch_kernel<<<(BB*2*SS)/8, blk, 0, stream>>>(resid, gi0, gi1, keep, slot, xe);
  ln_kernel<<<(EE*ROWS_E)/8, blk, 0, stream>>>(xe, ln_g, ln_b, xnb);

  // h = leaky(xn @ W1[e] + b1[e]) -> bf16, N=1365 padded to 1376 with zeros
  gemm_bf16_wmma<0,1,0,1><<<dim3(11,32,EE), blk, 0, stream>>>(
      xnb, W1b, b1, nullptr, hb, ROWS_E, HID, DD, DD, HIDP,
      (long long)ROWS_E*DD, (long long)DD*HID, HID, (long long)ROWS_E*HIDP);

  // ye = h @ W2[e] + b2[e] -> f32;  K padded to 1376 (valid B rows = 1365)
  gemm_bf16_wmma<0,0,0,0><<<dim3(4,32,EE), blk, 0, stream>>>(
      hb, W2b, b2, nullptr, ye, ROWS_E, DD, HIDP, HID, DD,
      (long long)ROWS_E*HIDP, (long long)HID*DD, DD, (long long)ROWS_E*DD);

  // out = resid + combined expert outputs; aux losses
  combine_kernel<<<NTOK/8, blk, 0, stream>>>(resid, ye, gi0, gi1, keep, slot, wcb, out);
  finalize_kernel<<<1, 32, 0, stream>>>(accum, out + (long long)NTOK*DD);
}
